// ARTEMISModel_22617297781308
// MI455X (gfx1250) — compile-verified, matrix operand order checked by reading, not run on hardware
//
#include <hip/hip_runtime.h>
#include <math.h>

#define HID 256
#define KSLOT 64
#define HEADS 4
#define DH 64
#define DECAYF 0.95f
#define THRF 2.0f
#define EPSF 1e-8f

typedef __attribute__((ext_vector_type(16))) _Float16 v16h;
typedef __attribute__((ext_vector_type(8)))  float    v8f;

// ---- LDS layout (bytes) ----
#define KV_STRIDE 257                         // floats; 257 % 64 == 1 -> conflict-free row access
#define U_STRIDE  264                         // halfs;  staggers A-fragment banks
#define OFF_K     0
#define OFF_V     (KV_STRIDE * KSLOT * 4)     // 65792
#define OFF_ST    0                           // storage overlaps K region (phases disjoint)
#define OFF_U     (2 * KV_STRIDE * KSLOT * 4) // 131584
#define OFF_SMALL (OFF_U + U_STRIDE * KSLOT * 2)  // 165376
#define SMALL_FLOATS (256*5 + 128 + 64 + 64 + 16)
#define SHMEM_BYTES  (OFF_SMALL + SMALL_FLOATS * 4)   // ~171.6 KB < 320 KB/WGP

// Swizzled f16 weight buffer in d_ws:
//   [mat(2)][ntile(16)][ks(8)][lane(32)][elem(16)] halfs -> 262144 bytes
#define WSW_HALFS (2 * 16 * 8 * 32 * 16)

__device__ __forceinline__ int fragK(int koff, int v) {
  return (v < 4) ? (koff + 2 * v) : (16 + koff + 2 * (v - 4));
}

// ---------- pre-kernel: swizzle Wk/Wv (f32, row-major) into WMMA-B fragment order (f16) ----------
__global__ __launch_bounds__(256)
void swizzle_weights(const float* __restrict__ in_w, _Float16* __restrict__ wsw) {
  const int idx = blockIdx.x * 256 + threadIdx.x;   // 0 .. WSW_HALFS-1
  const int e    = idx & 15;
  const int lane = (idx >> 4) & 31;
  const int ks   = (idx >> 9) & 7;
  const int nt   = (idx >> 12) & 15;
  const int mat  = idx >> 16;                       // 0 -> Wk, 1 -> Wv
  const int v    = e >> 1;
  const int p    = e & 1;
  const int koff = (lane & 16) ? 8 : 0;
  const int ncol = nt * 16 + (lane & 15);
  const int row  = (mat ? 2 * HID : HID) + ncol;    // row in in_proj_w
  const int kk   = ks * 32 + fragK(koff, v) + p;
  wsw[idx] = (_Float16)in_w[(size_t)row * HID + kk];
}

__device__ __forceinline__ void blockReduce256(float* buf, int tid) {
  for (int s = 128; s > 0; s >>= 1) {
    __syncthreads();
    if (tid < s) buf[tid] += buf[tid + s];
  }
  __syncthreads();
}

__global__ __launch_bounds__(256)
void artemis_fused(const float* __restrict__ storage,
                   const float* __restrict__ nmsg,
                   const float* __restrict__ w_ad1, const float* __restrict__ b_ad1,
                   const float* __restrict__ w_ad2, const float* __restrict__ b_ad2,
                   const float* __restrict__ w_im1, const float* __restrict__ b_im1,
                   const float* __restrict__ w_im2, const float* __restrict__ b_im2,
                   const float* __restrict__ in_w,  const float* __restrict__ in_b,
                   const float* __restrict__ out_w, const float* __restrict__ out_b,
                   const _Float16* __restrict__ wsw,
                   float* __restrict__ outUpd, float* __restrict__ outAgg)
{
  extern __shared__ char smemRaw[];
  float*    smK  = (float*)(smemRaw + OFF_K);
  float*    smV  = (float*)(smemRaw + OFF_V);
  float*    smSt = (float*)(smemRaw + OFF_ST);       // overlaps smK; storage dead before WMMA
  _Float16* smU  = (_Float16*)(smemRaw + OFF_U);
  float*    smSm = (float*)(smemRaw + OFF_SMALL);
  float* smNM   = smSm;            // 256
  float* smMean = smSm + 256;      // 256
  float* smQ    = smSm + 512;      // 256
  float* smTmp  = smSm + 768;      // 256 (g, then ctx_out)
  float* smS    = smSm + 1024;     // 256 (reduction scratch, then scores/attn)
  float* smH1   = smSm + 1280;     // 128
  float* smRow  = smSm + 1408;     // 64
  float* smDp   = smSm + 1472;     // 64
  float* smScal = smSm + 1536;     // 16

  const int tid = threadIdx.x;
  const int b   = blockIdx.x;
  const float* stg = storage + (size_t)b * (KSLOT * HID);

  // ---------- P0: stage inputs ----------
  smNM[tid] = nmsg[(size_t)b * HID + tid];
  if (tid < 64) smDp[tid] = powf(DECAYF, (float)tid);
  #pragma unroll
  for (int j = 0; j < 16; ++j) {
    int i4 = tid + j * 256;
    float4 v = ((const float4*)stg)[i4];
    *(float4*)(smSt + i4 * 4) = v;
  }
  __syncthreads();

  // ---------- P1: per-slot |.|-sum partials (4 threads per slot, bank-staggered) ----------
  {
    const int row = tid >> 2, seg = tid & 3;
    const float* rp = smSt + row * HID + seg * 64;
    float s = 0.f;
    for (int d0 = 0; d0 < 64; ++d0) { int d = (d0 + tid) & 63; s += fabsf(rp[d]); }
    smTmp[tid] = s;
  }
  __syncthreads();
  if (tid < 64)
    smRow[tid] = smTmp[4*tid] + smTmp[4*tid+1] + smTmp[4*tid+2] + smTmp[4*tid+3];
  __syncthreads();

  // ---------- P2: per-column stats + q-projection + anomaly MLP layer1 ----------
  {
    float msum = EPSF;
    for (int k = 0; k < KSLOT; ++k) msum += (smRow[k] > 0.f) ? 1.f : 0.f;
    float cs = 0.f;
    for (int k = 0; k < KSLOT; ++k) if (smRow[k] > 0.f) cs += smSt[k * HID + tid];
    const float mean = cs / msum;
    smMean[tid] = mean;
    float vs = 0.f;
    for (int k = 0; k < KSLOT; ++k) if (smRow[k] > 0.f) {
      float d = smSt[k * HID + tid] - mean; vs += d * d;
    }
    const float sd = sqrtf(vs / msum);
    const float z  = fabsf((smNM[tid] - mean) / (sd + EPSF));
    smS[tid] = (z > THRF) ? 1.f : 0.f;          // indicator for stat_anom

    float q = in_b[tid];
    const float* wq = in_w + (size_t)tid * HID;  // rows 0..255 = Wq
    for (int i = 0; i < HID; ++i) q = fmaf(smNM[i], wq[i], q);
    smQ[tid] = q;
  }
  if (tid < 128) {
    float h = b_ad1[tid];
    const float* wr = w_ad1 + (size_t)tid * HID;
    for (int i = 0; i < HID; ++i) h = fmaf(smNM[i], wr[i], h);
    smH1[tid] = fmaxf(h, 0.f);
  }
  blockReduce256(smS, tid);                     // sum of indicators
  if (tid == 0) smScal[1] = smS[0];
  __syncthreads();

  // ---------- P3: learned anomaly dot ----------
  smS[tid] = (tid < 128) ? smH1[tid] * w_ad2[tid] : 0.f;
  blockReduce256(smS, tid);
  if (tid == 0) smScal[2] = smS[0];

  // ---------- P4: importance gate ----------
  {
    float g = b_im1[tid];
    const float* wr = w_im1 + (size_t)tid * (2 * HID);
    for (int i = 0; i < HID; ++i) g = fmaf(smNM[i],   wr[i],       g);
    for (int i = 0; i < HID; ++i) g = fmaf(smMean[i], wr[HID + i], g);
    smTmp[tid] = fmaxf(g, 0.f);
  }
  __syncthreads();
  smS[tid] = smTmp[tid] * w_im2[tid];
  blockReduce256(smS, tid);
  if (tid == 0) {
    const float learned = 1.f / (1.f + expf(-(smScal[2] + b_ad2[0])));
    const float stat    = smScal[1] * (1.f / HID);
    const float anom    = 0.5f * stat + 0.5f * learned;
    const float x       = smS[0] + b_im2[0];
    const float sp      = (x > 20.f) ? x : log1pf(expf(x));
    smScal[4] = sp * (1.f + anom);              // imp
  }
  __syncthreads();

  // ---------- P5: materialize `updated` (f32 -> global, f16 -> LDS) ----------
  {
    const float imp = smScal[4];
    float* updg = outUpd + (size_t)b * (KSLOT * HID);
    #pragma unroll
    for (int j = 0; j < 16; ++j) {
      int i4 = tid + j * 256;
      int r  = i4 >> 6;                          // slot row 0..63
      int c4 = i4 & 63;                          // float4 column
      float x0, x1, x2, x3;
      if (r == 0) {
        const float* np = smNM + c4 * 4;
        x0 = np[0]*imp; x1 = np[1]*imp; x2 = np[2]*imp; x3 = np[3]*imp;
      } else {
        const float d = smDp[r - 1];
        const float* sp = smSt + (r - 1) * HID + c4 * 4;
        x0 = sp[0]*d; x1 = sp[1]*d; x2 = sp[2]*d; x3 = sp[3]*d;
      }
      float4 o; o.x = x0; o.y = x1; o.z = x2; o.w = x3;
      *(float4*)(updg + i4 * 4) = o;             // coalesced B128 store
      _Float16* up = smU + r * U_STRIDE + c4 * 4;
      up[0] = (_Float16)x0; up[1] = (_Float16)x1; up[2] = (_Float16)x2; up[3] = (_Float16)x3;
    }
  }
  __syncthreads();

  // ---------- P6: K/V projections via v_wmma_f32_16x16x32_f16 ----------
  // wave w: mat = w&1 (0->K, 1->V); n-tiles {w>>1, +4, +8, +12}
  // B fragments come pre-swizzled (f16) from d_ws: one aligned 32B load per lane per ks.
  {
    const int w    = tid >> 5;
    const int lane = tid & 31;
    const int mat  = w & 1;
    const int ngrp = w >> 1;
    const float* bias = in_b + (mat ? 2 * HID : HID);
    float* dst = mat ? smV : smK;
    const int M    = lane & 15;
    const int koff = (lane & 16) ? 8 : 0;

    for (int nt = ngrp; nt < 16; nt += 4) {
      // per-(mat,nt) swizzled block: 8 ks * 32 lanes * 16 halfs
      const _Float16* wblk = wsw + ((size_t)(mat * 16 + nt) * 8 * 32 * 16) + lane * 16;
      __builtin_prefetch(wblk, 0, 1);            // global_prefetch_b8
      v8f acc0 = {}, acc1 = {}, acc2 = {}, acc3 = {};
      for (int ks = 0; ks < 8; ++ks) {
        const int k0 = ks * 32;
        const v16h bf = *(const v16h*)(wblk + (size_t)ks * (32 * 16));  // 2x global_load_b128
        #pragma unroll
        for (int mi = 0; mi < 4; ++mi) {
          const _Float16* arow = smU + (mi * 16 + M) * U_STRIDE;
          v16h af;
          #pragma unroll
          for (int v = 0; v < 8; ++v) {
            const int kk = k0 + fragK(koff, v);
            af[2 * v]     = arow[kk];
            af[2 * v + 1] = arow[kk + 1];
          }
          v8f& acc = (mi == 0) ? acc0 : (mi == 1) ? acc1 : (mi == 2) ? acc2 : acc3;
          acc = __builtin_amdgcn_wmma_f32_16x16x32_f16(
                  false, af, false, bf, (short)0, acc, false, false);
        }
      }
      // D layout: lane col = lane&15; rows j (+8 for upper half-wave)
      const int col   = nt * 16 + (lane & 15);
      const int rbase = (lane & 16) ? 8 : 0;
      const float bv  = bias[col];
      #pragma unroll
      for (int j = 0; j < 8; ++j) {
        dst[(0 * 16 + rbase + j) * KV_STRIDE + col] = acc0[j] + bv;
        dst[(1 * 16 + rbase + j) * KV_STRIDE + col] = acc1[j] + bv;
        dst[(2 * 16 + rbase + j) * KV_STRIDE + col] = acc2[j] + bv;
        dst[(3 * 16 + rbase + j) * KV_STRIDE + col] = acc3[j] + bv;
      }
    }
  }
  __syncthreads();

  // ---------- P7: attention ----------
  {
    const int head = tid >> 6, r = tid & 63;
    const float* kr = smK + r * KV_STRIDE + head * DH;
    const float* qh = smQ + head * DH;
    float s = 0.f;
    for (int d = 0; d < DH; ++d) s = fmaf(qh[d], kr[d], s);
    smS[tid] = s * 0.125f;                       // 1/sqrt(64)
  }
  __syncthreads();
  if (tid < HEADS) {
    float* sp = smS + tid * 64;
    float mx = -3.4e38f;
    for (int r = 0; r < 64; ++r) mx = fmaxf(mx, sp[r]);
    float sum = 0.f;
    for (int r = 0; r < 64; ++r) { float e = expf(sp[r] - mx); sp[r] = e; sum += e; }
    const float inv = 1.f / sum;
    for (int r = 0; r < 64; ++r) sp[r] *= inv;
  }
  __syncthreads();
  {
    const int head = tid >> 6;
    const float* ap = smS + head * 64;
    float c = 0.f;
    for (int r = 0; r < 64; ++r) c = fmaf(ap[r], smV[r * KV_STRIDE + tid], c);
    smTmp[tid] = c;                              // ctx_out
  }
  __syncthreads();

  // ---------- P8: output projection ----------
  {
    float o = out_b[tid];
    const float* wr = out_w + (size_t)tid * HID;
    for (int i = 0; i < HID; ++i) o = fmaf(smTmp[i], wr[i], o);
    outAgg[(size_t)b * HID + tid] = o;
  }
}

extern "C" void kernel_launch(void* const* d_in, const int* in_sizes, int n_in,
                              void* d_out, int out_size, void* d_ws, size_t ws_size,
                              hipStream_t stream) {
  (void)n_in; (void)out_size; (void)ws_size;
  const float* storage = (const float*)d_in[0];
  const float* nmsg    = (const float*)d_in[1];
  const float* w_ad1   = (const float*)d_in[2];
  const float* b_ad1   = (const float*)d_in[3];
  const float* w_ad2   = (const float*)d_in[4];
  const float* b_ad2   = (const float*)d_in[5];
  const float* w_im1   = (const float*)d_in[6];
  const float* b_im1   = (const float*)d_in[7];
  const float* w_im2   = (const float*)d_in[8];
  const float* b_im2   = (const float*)d_in[9];
  const float* in_w    = (const float*)d_in[10];
  const float* in_b    = (const float*)d_in[11];
  const float* out_w   = (const float*)d_in[12];
  const float* out_b   = (const float*)d_in[13];

  const int B = in_sizes[1] / HID;               // 8192
  float* outUpd = (float*)d_out;                 // [B,64,256]
  float* outAgg = outUpd + (size_t)B * KSLOT * HID;  // [B,256]
  _Float16* wsw = (_Float16*)d_ws;               // 256 KB swizzled Wk/Wv (f16)

  swizzle_weights<<<WSW_HALFS / 256, 256, 0, stream>>>(in_w, wsw);
  artemis_fused<<<B, 256, SHMEM_BYTES, stream>>>(
      storage, nmsg, w_ad1, b_ad1, w_ad2, b_ad2, w_im1, b_im1, w_im2, b_im2,
      in_w, in_b, out_w, out_b, wsw, outUpd, outAgg);
}
